// GroupedQueryAttention_28183575396720
// MI455X (gfx1250) — matrix operand order repaired and checked by reading, not compiled, not run on hardware
//
#include <hip/hip_runtime.h>
#include <hip/hip_bf16.h>
#include <math.h>

typedef __attribute__((ext_vector_type(16))) __bf16 v16bf;
typedef __attribute__((ext_vector_type(8)))  float  v8f;

constexpr int BB   = 2;
constexpr int TT   = 2048;
constexpr int EE   = 2048;
constexpr int NHQ  = 32;
constexpr int NHKV = 8;
constexpr int HD   = 64;
constexpr int KVN  = NHKV * HD;   // 512
constexpr int GRP  = NHQ / NHKV;  // 4

// ---- helpers --------------------------------------------------------------
struct U32B { uint4 a, b; };

// Load a 16-half WMMA fragment as two 16-byte reads (-> ds_load_b128 x2)
__device__ __forceinline__ v16bf ldfrag(const __bf16* p0, const __bf16* p1) {
  U32B u;
  u.a = *(const uint4*)p0;
  u.b = *(const uint4*)p1;
  return __builtin_bit_cast(v16bf, u);
}

__device__ __forceinline__ unsigned pk2bf(float a, float b) {
  unsigned short ua = __builtin_bit_cast(unsigned short, (__bf16)a);
  unsigned short ub = __builtin_bit_cast(unsigned short, (__bf16)b);
  return (unsigned)ua | ((unsigned)ub << 16);
}

__device__ __forceinline__ float fget(const float4& v, int j) {
  return j == 0 ? v.x : j == 1 ? v.y : j == 2 ? v.z : v.w;
}

// CDNA5 async copy global -> LDS, 16B per lane, tracked by ASYNCcnt.
// Low 32 bits of a generic LDS pointer are the wave-relative LDS byte address.
__device__ __forceinline__ void async_copy_b128(void* lds, const void* g) {
  unsigned off = (unsigned)(uintptr_t)lds;
  asm volatile("global_load_async_to_lds_b128 %0, %1, off"
               :: "v"(off), "v"(g) : "memory");
}
__device__ __forceinline__ void wait_async0() {
  asm volatile("s_wait_asynccnt 0x0" ::: "memory");
}

// ---------------------------------------------------------------------------
// GEMM: C[M,N] = A[M,K] @ W[K,N] + bias[N]   (fp32 in/out, bf16 WMMA compute)
// Block 128x128, 8 waves (4x2), each wave 32x64 = 2x4 WMMA tiles, K-step 32.
// Double-buffered LDS: next tile's global loads issued before computing on
// the current buffer. A staged [m][k]; B staged TRANSPOSED [n][k].
// ---------------------------------------------------------------------------
constexpr int GBM = 128, GBN = 128, GBK = 32;
constexpr int GLDA  = GBK + 8;   // 40 halves, row stride 80B (16B multiple)
constexpr int GLDBT = GBK + 8;   // 40 halves

__global__ __launch_bounds__(256)
void gemm_bias_bf16wmma(const float* __restrict__ A, const float* __restrict__ W,
                        const float* __restrict__ bias, float* __restrict__ C,
                        int M, int N, int K)
{
  __shared__ alignas(16) __bf16 sA [2][GBM * GLDA];
  __shared__ alignas(16) __bf16 sBT[2][GBN * GLDBT];
  const int tid  = threadIdx.x;
  const int lane = tid & 31;
  const int wave = tid >> 5;
  const int wm = wave >> 1, wn = wave & 1;   // wave tile: rows wm*32, cols wn*64
  const int bm = blockIdx.y * GBM;
  const int bn = blockIdx.x * GBN;
  const int kbA = (lane >> 4) * 8;    // A-fragment K base (ISA 16-bit A layout)
  const int kbB = (lane >> 4) * 8;    // B-fragment K base (ISA 16-bit B layout)
  const int brg = (tid >> 5) * 4;     // B micro-tile k rows 0..28
  const int bcg = (tid & 31) * 4;     // B micro-tile n cols 0..124

  v8f acc[2][4] = {};
  float4 va[4], vb[4];

  auto load_tile = [&](int k0) {
#pragma unroll
    for (int r = 0; r < 4; ++r) {
      int c4 = tid + r * 256;
      va[r] = *(const float4*)(A + (size_t)(bm + (c4 >> 3)) * K + (k0 + ((c4 & 7) << 2)));
    }
#pragma unroll
    for (int t = 0; t < 4; ++t)
      vb[t] = *(const float4*)(W + (size_t)(k0 + brg + t) * N + (bn + bcg));
  };
  auto store_tile = [&](int buf) {
#pragma unroll
    for (int r = 0; r < 4; ++r) {
      int c4 = tid + r * 256;
      uint2 w;
      w.x = pk2bf(va[r].x, va[r].y);
      w.y = pk2bf(va[r].z, va[r].w);
      *(uint2*)(&sA[buf][(c4 >> 3) * GLDA + ((c4 & 7) << 2)]) = w;
    }
#pragma unroll
    for (int j2 = 0; j2 < 4; ++j2) {            // transpose 4x4 micro-tile
      uint2 w;
      w.x = pk2bf(fget(vb[0], j2), fget(vb[1], j2));
      w.y = pk2bf(fget(vb[2], j2), fget(vb[3], j2));
      *(uint2*)(&sBT[buf][(bcg + j2) * GLDBT + brg]) = w;
    }
  };

  load_tile(0);
  store_tile(0);
  __syncthreads();

  const int nk = K / GBK;
  for (int kt = 0; kt < nk; ++kt) {
    const int cur = kt & 1;
    if (kt + 1 < nk) load_tile((kt + 1) * GBK);      // overlap with WMMA below
    if (kt + 2 < nk) {                               // L2 hint two tiles ahead
      __builtin_prefetch(A + (size_t)(bm + (tid >> 1)) * K + (kt + 2) * GBK, 0, 1);
      __builtin_prefetch(W + (size_t)((kt + 2) * GBK + (tid >> 3)) * N + bn, 0, 1);
    }

    const __bf16* sAc = sA[cur];
    const __bf16* sBc = sBT[cur];
    v16bf af[2], bfr[4];
#pragma unroll
    for (int mt = 0; mt < 2; ++mt) {
      const __bf16* base = sAc + (wm * 32 + mt * 16 + (lane & 15)) * GLDA + kbA;
      af[mt] = ldfrag(base, base + 16);
    }
#pragma unroll
    for (int nt = 0; nt < 4; ++nt) {
      const __bf16* base = sBc + (wn * 64 + nt * 16 + (lane & 15)) * GLDBT + kbB;
      bfr[nt] = ldfrag(base, base + 16);
    }
#pragma unroll
    for (int mt = 0; mt < 2; ++mt)
#pragma unroll
      for (int nt = 0; nt < 4; ++nt)
        acc[mt][nt] = __builtin_amdgcn_wmma_f32_16x16x32_bf16(
            false, af[mt], false, bfr[nt], (short)0, acc[mt][nt], false, false);

    if (kt + 1 < nk) store_tile(cur ^ 1);
    __syncthreads();
  }

#pragma unroll
  for (int mt = 0; mt < 2; ++mt)
#pragma unroll
    for (int nt = 0; nt < 4; ++nt) {
      const int col = bn + wn * 64 + nt * 16 + (lane & 15);
      const float bv = bias[col];
#pragma unroll
      for (int i = 0; i < 8; ++i) {
        const int row = bm + wm * 32 + mt * 16 + (lane >> 4) * 8 + i;
        C[(size_t)row * N + col] = acc[mt][nt][i] + bv;
      }
    }
}

// ---------------------------------------------------------------------------
// RoPE + layout change: fp32 [B,T,H*D] -> bf16 [B,H,T,D], positions 1..T.
// ---------------------------------------------------------------------------
__global__ void rope_pack(const float* __restrict__ in, unsigned short* __restrict__ outu, int H)
{
  __bf16* out = (__bf16*)outu;
  const int idx  = blockIdx.x * blockDim.x + threadIdx.x;
  const int half = HD / 2;
  const int i = idx % half;
  const int h = (idx / half) % H;
  const int t = (idx / (half * H)) % TT;
  const int b = idx / (half * H * TT);
  const size_t ib = ((size_t)(b * TT + t)) * (size_t)(H * HD) + h * HD;
  const float x1 = in[ib + i];
  const float x2 = in[ib + i + half];
  const float theta = (float)(t + 1) * __powf(10000.f, (-2.f * (float)i) / (float)HD);
  const float c = __cosf(theta), s = __sinf(theta);
  const size_t ob = (((size_t)b * H + h) * TT + t) * (size_t)HD;
  out[ob + i]        = (__bf16)(x1 * c - x2 * s);
  out[ob + i + half] = (__bf16)(x2 * c + x1 * s);
}

// V: fp32 [B,T,HKV*D] -> bf16 [B,HKV,T,D]
__global__ void v_pack(const float* __restrict__ in, unsigned short* __restrict__ outu)
{
  __bf16* out = (__bf16*)outu;
  const int idx = blockIdx.x * blockDim.x + threadIdx.x;
  const int d = idx % HD;
  const int h = (idx / HD) % NHKV;
  const int t = (idx / (HD * NHKV)) % TT;
  const int b = idx / (HD * NHKV * TT);
  out[(((size_t)b * NHKV + h) * TT + t) * HD + d] =
      (__bf16)in[((size_t)(b * TT + t)) * KVN + h * HD + d];
}

// ---------------------------------------------------------------------------
// Causal flash attention, GQA. Grid (T/128, B*HQ), 8 waves, 16 q-rows/wave.
// Double-buffered K/V tiles. K tile copied with GLOBAL_LOAD_ASYNC_TO_LDS_B128
// (ASYNCcnt) issued one block ahead; V transposed through registers.
// ---------------------------------------------------------------------------
constexpr int ABM  = 128;      // q rows per block
constexpr int ABN  = 64;       // keys per iteration
constexpr int ALDK = HD + 8;   // 72 halves, 144B stride (16B multiple)
constexpr int ALDV = ABN + 8;  // 72 halves
constexpr int ALDP = ABN + 8;  // 72 halves

__global__ __launch_bounds__(256)
void attn_fwd(const unsigned short* __restrict__ Qu, const unsigned short* __restrict__ Ku,
              const unsigned short* __restrict__ Vu, float* __restrict__ Y)
{
  const __bf16* Qg = (const __bf16*)Qu;
  const __bf16* Kg = (const __bf16*)Ku;
  const __bf16* Vg = (const __bf16*)Vu;

  __shared__ alignas(16) __bf16 sK [2][ABN * ALDK];   // [key][d]
  __shared__ alignas(16) __bf16 sVT[2][HD  * ALDV];   // [d][key]
  __shared__ alignas(16) __bf16 sP [8 * 16 * ALDP];

  const int tid = threadIdx.x, lane = tid & 31, wave = tid >> 5;
  const int qblk = blockIdx.x;
  const int bh   = blockIdx.y;
  const int b = bh / NHQ, h = bh % NHQ, kvh = h / GRP;
  const __bf16* Qh = Qg + ((size_t)(b * NHQ  + h))   * TT * HD;
  const __bf16* Kh = Kg + ((size_t)(b * NHKV + kvh)) * TT * HD;
  const __bf16* Vh = Vg + ((size_t)(b * NHKV + kvh)) * TT * HD;
  const int qrow0 = qblk * ABM + wave * 16;
  const int kbA = (lane >> 4) * 8;
  const int kbB = (lane >> 4) * 8;
  const int vrg = (tid >> 4) * 4;   // V micro-tile key rows 0..60
  const int vcg = (tid & 15) * 4;   // V micro-tile d cols  0..60

  // Q fragments straight from global (two contiguous 16B runs per chunk)
  v16bf qf[2];
  {
    const __bf16* qp = Qh + (size_t)(qrow0 + (lane & 15)) * HD;
#pragma unroll
    for (int c = 0; c < 2; ++c)
      qf[c] = ldfrag(qp + c * 32 + kbA, qp + c * 32 + kbA + 16);
  }

  uint2 vld[4];
  auto issueK = [&](int j0, int buf) {   // async copy: 2 x 16B per thread
#pragma unroll
    for (int r = 0; r < 2; ++r) {
      int c8 = tid + r * 256;
      async_copy_b128(&sK[buf][(c8 >> 3) * ALDK + ((c8 & 7) << 3)],
                      Kh + (size_t)(j0 + (c8 >> 3)) * HD + ((c8 & 7) << 3));
    }
  };
  auto loadV = [&](int j0) {
#pragma unroll
    for (int t = 0; t < 4; ++t)
      vld[t] = *(const uint2*)(Vh + (size_t)(j0 + vrg + t) * HD + vcg);
  };
  auto storeV = [&](int buf) {           // transpose bf16 4x4 in registers
#pragma unroll
    for (int j2 = 0; j2 < 4; ++j2) {
      unsigned h0 = ((j2 < 2 ? vld[0].x : vld[0].y) >> ((j2 & 1) * 16)) & 0xffffu;
      unsigned h1 = ((j2 < 2 ? vld[1].x : vld[1].y) >> ((j2 & 1) * 16)) & 0xffffu;
      unsigned h2 = ((j2 < 2 ? vld[2].x : vld[2].y) >> ((j2 & 1) * 16)) & 0xffffu;
      unsigned h3 = ((j2 < 2 ? vld[3].x : vld[3].y) >> ((j2 & 1) * 16)) & 0xffffu;
      uint2 w;
      w.x = h0 | (h1 << 16);
      w.y = h2 | (h3 << 16);
      *(uint2*)(&sVT[buf][(vcg + j2) * ALDV + vrg]) = w;
    }
  };

  v8f o[4] = {};
  float mrun[8], lrun[8];
#pragma unroll
  for (int i = 0; i < 8; ++i) { mrun[i] = -3.0e30f; lrun[i] = 0.f; }

  issueK(0, 0);
  loadV(0);
  storeV(0);
  wait_async0();
  __syncthreads();

  const int nkb = ((qblk + 1) * ABM) / ABN;
  for (int kb = 0; kb < nkb; ++kb) {
    const int cur = kb & 1;
    const int j0  = kb * ABN;
    const bool more = (kb + 1) < nkb;
    if (more) { issueK(j0 + ABN, cur ^ 1); loadV(j0 + ABN); }

    const __bf16* sKc = sK[cur];
    const __bf16* sVc = sVT[cur];

    // ---- S = Q K^T ----
    v8f s[4] = {};
#pragma unroll
    for (int nt = 0; nt < 4; ++nt) {
      const int key = nt * 16 + (lane & 15);
#pragma unroll
      for (int c = 0; c < 2; ++c) {
        const __bf16* base = sKc + key * ALDK + c * 32 + kbB;
        v16bf kf = ldfrag(base, base + 16);
        s[nt] = __builtin_amdgcn_wmma_f32_16x16x32_bf16(
            false, qf[c], false, kf, (short)0, s[nt], false, false);
      }
    }

    // ---- scale + causal mask + online softmax ----
    const float scale = 0.125f;   // 1/sqrt(64)
    float mnew[8];
#pragma unroll
    for (int i = 0; i < 8; ++i) mnew[i] = mrun[i];
#pragma unroll
    for (int nt = 0; nt < 4; ++nt) {
      const int col = j0 + nt * 16 + (lane & 15);
#pragma unroll
      for (int i = 0; i < 8; ++i) {
        const int row = qrow0 + (lane >> 4) * 8 + i;
        float v = s[nt][i] * scale;
        if (col > row) v = -3.0e30f;
        s[nt][i] = v;
        mnew[i] = fmaxf(mnew[i], v);
      }
    }
#pragma unroll
    for (int i = 0; i < 8; ++i) {
      float v = mnew[i];
      v = fmaxf(v, __shfl_xor(v, 1, 32));
      v = fmaxf(v, __shfl_xor(v, 2, 32));
      v = fmaxf(v, __shfl_xor(v, 4, 32));
      v = fmaxf(v, __shfl_xor(v, 8, 32));
      mnew[i] = v;
    }
    float alpha[8], lsum[8];
#pragma unroll
    for (int i = 0; i < 8; ++i) {
      alpha[i] = __expf(mrun[i] - mnew[i]);
      mrun[i]  = mnew[i];
      lrun[i] *= alpha[i];
      lsum[i]  = 0.f;
    }
#pragma unroll
    for (int nt = 0; nt < 4; ++nt)
#pragma unroll
      for (int i = 0; i < 8; ++i) {
        o[nt][i] *= alpha[i];
        float p = __expf(s[nt][i] - mnew[i]);
        s[nt][i] = p;
        lsum[i] += p;
      }
#pragma unroll
    for (int i = 0; i < 8; ++i) {
      float v = lsum[i];
      v += __shfl_xor(v, 1, 32);
      v += __shfl_xor(v, 2, 32);
      v += __shfl_xor(v, 4, 32);
      v += __shfl_xor(v, 8, 32);
      lrun[i] += v;
    }

    // ---- stage P (C-layout -> row-major strip) ----
    __bf16* pst = sP + wave * 16 * ALDP;
#pragma unroll
    for (int nt = 0; nt < 4; ++nt) {
      const int col = nt * 16 + (lane & 15);
#pragma unroll
      for (int i = 0; i < 8; ++i)
        pst[((lane >> 4) * 8 + i) * ALDP + col] = (__bf16)s[nt][i];
    }

    // ---- O += P @ V ----
#pragma unroll
    for (int nt = 0; nt < 4; ++nt) {
#pragma unroll
      for (int c = 0; c < 2; ++c) {
        const __bf16* pb = pst + (lane & 15) * ALDP + c * 32 + kbA;
        const __bf16* vbp = sVc + (nt * 16 + (lane & 15)) * ALDV + c * 32 + kbB;
        v16bf pf = ldfrag(pb, pb + 16);
        v16bf vf = ldfrag(vbp, vbp + 16);
        o[nt] = __builtin_amdgcn_wmma_f32_16x16x32_bf16(
            false, pf, false, vf, (short)0, o[nt], false, false);
      }
    }

    if (more) storeV(cur ^ 1);
    wait_async0();        // this wave's async K writes landed in LDS
    __syncthreads();
  }

  // ---- normalize + store fp32 [B,T,E] ----
#pragma unroll
  for (int nt = 0; nt < 4; ++nt) {
    const int d = nt * 16 + (lane & 15);
#pragma unroll
    for (int i = 0; i < 8; ++i) {
      const int row = qrow0 + (lane >> 4) * 8 + i;
      Y[((size_t)(b * TT + row)) * EE + h * HD + d] = o[nt][i] / lrun[i];
    }
  }
}

// ---------------------------------------------------------------------------
extern "C" void kernel_launch(void* const* d_in, const int* in_sizes, int n_in,
                              void* d_out, int out_size, void* d_ws, size_t ws_size,
                              hipStream_t stream)
{
  const float* x  = (const float*)d_in[0];
  const float* Wq = (const float*)d_in[1];
  const float* bq = (const float*)d_in[2];
  const float* Wk = (const float*)d_in[3];
  const float* bk = (const float*)d_in[4];
  const float* Wv = (const float*)d_in[5];
  const float* bv = (const float*)d_in[6];
  const float* Wo = (const float*)d_in[7];
  const float* bo = (const float*)d_in[8];
  float* out = (float*)d_out;

  char* ws = (char*)d_ws;
  const size_t M = (size_t)BB * TT;  // 4096 rows
  float* qproj = (float*)ws;                                                    // 32 MB
  float* kproj = (float*)(ws + M * EE * 4);                                     // 8 MB
  float* vproj = (float*)(ws + M * EE * 4 + M * KVN * 4);                       // 8 MB
  unsigned short* qb16 = (unsigned short*)(ws + M * EE * 4 + 2 * M * KVN * 4);  // 16 MB
  unsigned short* kb16 = (unsigned short*)((char*)qb16 + M * EE * 2);           // 4 MB
  unsigned short* vb16 = (unsigned short*)((char*)kb16 + M * KVN * 2);          // 4 MB
  float* yattn = qproj;   // qproj fully consumed by rope_pack before attn_fwd

  dim3 blk(256);
  gemm_bias_bf16wmma<<<dim3(EE / GBN,  M / GBM), blk, 0, stream>>>(x, Wq, bq, qproj, (int)M, EE,  EE);
  gemm_bias_bf16wmma<<<dim3(KVN / GBN, M / GBM), blk, 0, stream>>>(x, Wk, bk, kproj, (int)M, KVN, EE);
  gemm_bias_bf16wmma<<<dim3(KVN / GBN, M / GBM), blk, 0, stream>>>(x, Wv, bv, vproj, (int)M, KVN, EE);
  rope_pack<<<(BB * TT * NHQ  * (HD / 2)) / 256, blk, 0, stream>>>(qproj, qb16, NHQ);
  rope_pack<<<(BB * TT * NHKV * (HD / 2)) / 256, blk, 0, stream>>>(kproj, kb16, NHKV);
  v_pack<<<(BB * TT * KVN) / 256, blk, 0, stream>>>(vproj, vb16);
  attn_fwd<<<dim3(TT / ABM, BB * NHQ), blk, 0, stream>>>(qb16, kb16, vb16, yattn);
  gemm_bias_bf16wmma<<<dim3(EE / GBN, M / GBM), blk, 0, stream>>>(yattn, Wo, bo, out, (int)M, EE, EE);
}
